// VoteFusionModuleV3_26688926777519
// MI455X (gfx1250) — compile-verified, matrix operand order checked by reading, not run on hardware
//
#include <hip/hip_runtime.h>
#include <hip/hip_bf16.h>

typedef __attribute__((ext_vector_type(16))) _Float16 v16h;
typedef __attribute__((ext_vector_type(8)))  float    v8f;

#define N_PCDp 5000
#define N_RGBp 15000
#define N_ALLp 20000
#define N_ANCH 4096

// ---------------------------------------------------------------------------
// Prep: convert weights to f16 (with padding), concat xyz
// ---------------------------------------------------------------------------
__global__ __launch_bounds__(256) void prep_weights_kernel(
    const float* saW1, const float* saW2, const float* saW3,
    const float* fpW1, const float* fpW2,
    _Float16* w1h, _Float16* w2h, _Float16* w3h,
    _Float16* f1h, _Float16* f2h)
{
  int i = blockIdx.x * 256 + threadIdx.x;
  if (i < 128 * 192) {  // sa_W1: 128 x 163 -> 128 x 192 (zero pad)
    int o = i / 192, c = i % 192;
    w1h[i] = (_Float16)(c < 163 ? saW1[o * 163 + c] : 0.0f);
  }
  if (i < 128 * 128) {
    w2h[i] = (_Float16)saW2[i];
    w3h[i] = (_Float16)saW3[i];
  }
  if (i < 128 * 160) {  // fp_W1: keep first 160 of 288 cols (rest multiply zeros)
    int o = i / 160, c = i % 160;
    f1h[i] = (_Float16)fpW1[o * 288 + c];
  }
  if (i < 32 * 128) f2h[i] = (_Float16)fpW2[i];
}

__global__ __launch_bounds__(256) void concat_xyz_kernel(
    const float* pcd, const float* rgb, float* xyz)
{
  int i = blockIdx.x * 256 + threadIdx.x;
  if (i < N_PCDp * 3) xyz[i] = pcd[i];
  if (i < N_RGBp * 3) xyz[N_PCDp * 3 + i] = rgb[i];
}

// ---------------------------------------------------------------------------
// FPS: single persistent block, dists + xyz strips in registers
// ---------------------------------------------------------------------------
__global__ __launch_bounds__(1024) void fps_kernel(const float* xyz, float* newxyz)
{
  __shared__ float redv[1024];
  __shared__ int   redi[1024];
  __shared__ int   scur;
  int t = threadIdx.x;
  float px[20], py[20], pz[20], dl[20];
#pragma unroll
  for (int k = 0; k < 20; ++k) {
    int i = t + k * 1024;
    if (i < N_ALLp) { px[k] = xyz[i*3]; py[k] = xyz[i*3+1]; pz[k] = xyz[i*3+2]; }
    else            { px[k] = 0.f; py[k] = 0.f; pz[k] = 0.f; }
    dl[k] = 1e10f;
  }
  int cur = 0;
  if (t == 0) { newxyz[0] = xyz[0]; newxyz[1] = xyz[1]; newxyz[2] = xyz[2]; }
  for (int j = 1; j < N_ANCH; ++j) {
    float cx = xyz[cur*3], cy = xyz[cur*3+1], cz = xyz[cur*3+2];
    float bv = -1.0f; int bi = 0x7fffffff;
#pragma unroll
    for (int k = 0; k < 20; ++k) {
      int i = t + k * 1024;
      float dx = px[k]-cx, dy = py[k]-cy, dz = pz[k]-cz;
      float d = dx*dx + dy*dy + dz*dz;
      float nd = fminf(dl[k], d);
      dl[k] = nd;
      if (i < N_ALLp && nd > bv) { bv = nd; bi = i; }
    }
    redv[t] = bv; redi[t] = bi;
    __syncthreads();
    for (int s = 512; s > 0; s >>= 1) {
      if (t < s) {
        float ov = redv[t+s]; int oi = redi[t+s];
        if (ov > redv[t] || (ov == redv[t] && oi < redi[t])) { redv[t] = ov; redi[t] = oi; }
      }
      __syncthreads();
    }
    if (t == 0) {
      int c = redi[0];
      scur = c;
      newxyz[j*3+0] = xyz[c*3+0];
      newxyz[j*3+1] = xyz[c*3+1];
      newxyz[j*3+2] = xyz[c*3+2];
    }
    __syncthreads();
    cur = scur;
  }
}

// ---------------------------------------------------------------------------
// Ball query: one wave per center, ordered index scan (matches top_k semantics)
// ---------------------------------------------------------------------------
__global__ __launch_bounds__(256) void ball_query_kernel(
    const float* pts, const float* ctr, int n, int m, float r2, int nsample, int* out)
{
  int gw   = (blockIdx.x * 256 + threadIdx.x) >> 5;
  int lane = threadIdx.x & 31;
  if (gw >= m) return;
  float cx = ctr[gw*3], cy = ctr[gw*3+1], cz = ctr[gw*3+2];
  int count = 0, firstIdx = 0;
  bool have = false;
  for (int base = 0; base < n && count < nsample; base += 32) {
    int i = base + lane;
    bool inball = false;
    if (i < n) {
      float dx = cx - pts[i*3], dy = cy - pts[i*3+1], dz = cz - pts[i*3+2];
      inball = (dx*dx + dy*dy + dz*dz) < r2;
    }
    unsigned mask = __builtin_amdgcn_ballot_w32(inball);
    if (!have && mask) { firstIdx = base + __builtin_ctz(mask); have = true; }
    if (inball) {
      int pos = count + __popc(mask & ((1u << lane) - 1u));
      if (pos < nsample) out[gw * nsample + pos] = i;
    }
    count += __popc(mask);
  }
  int filled = count < nsample ? count : nsample;
  for (int s = lane; s < nsample; s += 32)
    if (s >= filled) out[gw * nsample + s] = firstIdx;
}

// ---------------------------------------------------------------------------
// Wave-level WMMA layer helper: out[32x128] = relu(A[32xK] @ W^T + b)
// A in LDS (f16, row-major, stride lda), W in global (f16, 128 rows x lda)
// ---------------------------------------------------------------------------
__device__ __forceinline__ void wave_layer(
    const _Float16* A, int lda, int kChunks,
    const _Float16* W, const float* bias,
    int wave, int lane, _Float16* outH, float* outF)
{
  const int n0    = wave * 16;
  const int l15   = lane & 15;
  const int half8 = (lane >> 4) * 8;
  for (int mt = 0; mt < 2; ++mt) {
    v8f acc = {};
    const int mrow = mt * 16 + l15;
    for (int kc = 0; kc < kChunks; ++kc) {
      v16h af, bf;
      const _Float16* Ap = A + mrow * lda + kc * 32 + half8;
      const _Float16* Wp = W + (n0 + l15) * lda + kc * 32 + half8;
#pragma unroll
      for (int h = 0; h < 8; ++h)  { af[h] = Ap[h];     bf[h] = Wp[h]; }
#pragma unroll
      for (int h = 8; h < 16; ++h) { af[h] = Ap[h + 8]; bf[h] = Wp[h + 8]; }
      acc = __builtin_amdgcn_wmma_f32_16x16x32_f16(false, af, false, bf,
                                                   (short)0, acc, false, false);
    }
    float bv = bias[n0 + l15];
#pragma unroll
    for (int r = 0; r < 8; ++r) {
      float v = acc[r] + bv;
      v = v > 0.f ? v : 0.f;
      int m = mt * 16 + half8 + r;
      int n = n0 + l15;
      if (outH) outH[m * 128 + n] = (_Float16)v;
      if (outF) outF[m * 128 + n] = v;
    }
  }
}

// ---------------------------------------------------------------------------
// SA MLP: one anchor per block (256 threads = 8 waves), 3 WMMA layers + maxpool
// ---------------------------------------------------------------------------
__global__ __launch_bounds__(256) void sa_mlp_kernel(
    const float* xyz, const float* newxyz, const int* nbsa,
    const float* pcdf, const float* rgbf,
    const _Float16* W1, const _Float16* W2, const _Float16* W3,
    const float* b1, const float* b2, const float* b3,
    float* safeat)
{
  __shared__ _Float16 bufA[32 * 192];
  __shared__ _Float16 bufB[32 * 128];
  __shared__ float    bufC[32 * 128];
  int a = blockIdx.x;
  int t = threadIdx.x;
  float nx = newxyz[a*3], ny = newxyz[a*3+1], nz = newxyz[a*3+2];
  for (int e = t; e < 32 * 192; e += 256) {
    int s = e / 192, c = e % 192;
    int p = nbsa[a * 32 + s];
    float v = 0.f;
    if (c < 3) {
      float ctr = (c == 0) ? nx : ((c == 1) ? ny : nz);
      v = (xyz[p * 3 + c] - ctr) / 0.1f;
    } else if (c < 163) {
      int cc = c - 3;
      if (p < N_PCDp) { if (cc < 32) v = pcdf[p * 32 + cc]; }
      else            { if (cc >= 32) v = rgbf[(cc - 32) * N_RGBp + (p - N_PCDp)]; }
    }
    bufA[e] = (_Float16)v;
  }
  __syncthreads();
  int wave = t >> 5, lane = t & 31;
  wave_layer(bufA, 192, 6, W1, b1, wave, lane, bufB, nullptr);
  __syncthreads();
  wave_layer(bufB, 128, 4, W2, b2, wave, lane, bufA, nullptr);  // reuse bufA, stride 128
  __syncthreads();
  wave_layer(bufA, 128, 4, W3, b3, wave, lane, nullptr, bufC);
  __syncthreads();
  if (t < 128) {
    float m = bufC[t];
#pragma unroll 4
    for (int s = 1; s < 32; ++s) m = fmaxf(m, bufC[s * 128 + t]);
    safeat[a * 128 + t] = m;
  }
}

// ---------------------------------------------------------------------------
// FP interpolation: one wave per pcd point; 3-NN over anchors + build A matrix
// Afp row = [interp(128), pcd_features(32)] as f16 (rows 5000..5007 zero)
// ---------------------------------------------------------------------------
__global__ __launch_bounds__(256) void fp_interp_kernel(
    const float* pcd_xyz, const float* pcd_feat,
    const float* newxyz, const float* safeat, _Float16* Afp)
{
  int p    = (blockIdx.x * 256 + threadIdx.x) >> 5;
  int lane = threadIdx.x & 31;
  if (p >= 5008) return;
  if (p >= N_PCDp) {
    for (int c = lane; c < 160; c += 32) Afp[p * 160 + c] = (_Float16)0.f;
    return;
  }
  float px = pcd_xyz[p*3], py = pcd_xyz[p*3+1], pz = pcd_xyz[p*3+2];
  float d0 = 3.4e38f, d1 = 3.4e38f, d2 = 3.4e38f;
  int   i0 = 0x7fffffff, i1 = 0x7fffffff, i2 = 0x7fffffff;
  for (int a = lane; a < N_ANCH; a += 32) {
    float dx = newxyz[a*3]-px, dy = newxyz[a*3+1]-py, dz = newxyz[a*3+2]-pz;
    float d = dx*dx + dy*dy + dz*dz;
    if (d < d2) {
      if (d < d1) {
        d2 = d1; i2 = i1;
        if (d < d0) { d1 = d0; i1 = i0; d0 = d; i0 = a; }
        else        { d1 = d;  i1 = a; }
      } else { d2 = d; i2 = a; }
    }
  }
  float cd[3] = { d0, d1, d2 };
  int   ci[3] = { i0, i1, i2 };
  int k = 0;
  float rd[3]; int ri[3];
  for (int r = 0; r < 3; ++r) {
    float md = (k < 3) ? cd[k] : 3.4e38f;
    int   mi = (k < 3) ? ci[k] : 0x7fffffff;
    for (int off = 16; off > 0; off >>= 1) {
      float od = __shfl_xor(md, off);
      int   oi = __shfl_xor(mi, off);
      if (od < md || (od == md && oi < mi)) { md = od; mi = oi; }
    }
    rd[r] = md; ri[r] = mi;
    if (k < 3 && ci[k] == mi) k++;
  }
  float w0 = 1.f / (rd[0] + 1e-8f);
  float w1 = 1.f / (rd[1] + 1e-8f);
  float w2 = 1.f / (rd[2] + 1e-8f);
  float wsum = w0 + w1 + w2;
  w0 /= wsum; w1 /= wsum; w2 /= wsum;
  for (int c = lane; c < 128; c += 32) {
    float v = w0 * safeat[ri[0]*128 + c] + w1 * safeat[ri[1]*128 + c]
            + w2 * safeat[ri[2]*128 + c];
    Afp[p * 160 + c] = (_Float16)v;
  }
  Afp[p * 160 + 128 + lane] = (_Float16)pcd_feat[p * 32 + lane];
}

// ---------------------------------------------------------------------------
// Generic WMMA GEMM: one wave per 16x16 tile; out = relu(A @ W^T + b)
// ---------------------------------------------------------------------------
__global__ __launch_bounds__(256) void fp_gemm_kernel(
    const _Float16* A, const _Float16* W, const float* bias,
    int lda, int kChunks, int Ntiles, int nWaves,
    _Float16* outH, float* outF, int ldo)
{
  int gw   = (blockIdx.x * 256 + threadIdx.x) >> 5;
  int lane = threadIdx.x & 31;
  if (gw >= nWaves) return;
  int mt = gw / Ntiles, nt = gw % Ntiles;
  int l15 = lane & 15, half8 = (lane >> 4) * 8;
  int mrow = mt * 16 + l15;
  int ncol = nt * 16 + l15;
  v8f acc = {};
  for (int kc = 0; kc < kChunks; ++kc) {
    v16h af, bf;
    const _Float16* Ap = A + mrow * lda + kc * 32 + half8;
    const _Float16* Wp = W + ncol * lda + kc * 32 + half8;
#pragma unroll
    for (int h = 0; h < 8; ++h)  { af[h] = Ap[h];     bf[h] = Wp[h]; }
#pragma unroll
    for (int h = 8; h < 16; ++h) { af[h] = Ap[h + 8]; bf[h] = Wp[h + 8]; }
    acc = __builtin_amdgcn_wmma_f32_16x16x32_f16(false, af, false, bf,
                                                 (short)0, acc, false, false);
  }
  float bv = bias[ncol];
#pragma unroll
  for (int r = 0; r < 8; ++r) {
    float v = acc[r] + bv;
    v = v > 0.f ? v : 0.f;
    int m = mt * 16 + half8 + r;
    int n = nt * 16 + l15;
    if (outH) outH[m * ldo + n] = (_Float16)v;
    if (outF) outF[m * ldo + n] = v;
  }
}

// ---------------------------------------------------------------------------
// Global max of fp_feat (two-stage)
// ---------------------------------------------------------------------------
__global__ __launch_bounds__(256) void max_part_kernel(const float* x, int n, float* part)
{
  __shared__ float sm[256];
  int t = threadIdx.x;
  float m = -3.4e38f;
  for (int i = blockIdx.x * 256 + t; i < n; i += gridDim.x * 256) m = fmaxf(m, x[i]);
  sm[t] = m; __syncthreads();
  for (int s = 128; s > 0; s >>= 1) { if (t < s) sm[t] = fmaxf(sm[t], sm[t + s]); __syncthreads(); }
  if (t == 0) part[blockIdx.x] = sm[0];
}

__global__ __launch_bounds__(256) void max_final_kernel(const float* part, int n, float* gmax)
{
  __shared__ float sm[256];
  int t = threadIdx.x;
  sm[t] = (t < n) ? part[t] : -3.4e38f;
  __syncthreads();
  for (int s = 128; s > 0; s >>= 1) { if (t < s) sm[t] = fmaxf(sm[t], sm[t + s]); __syncthreads(); }
  if (t == 0) gmax[0] = sm[0];
}

// ---------------------------------------------------------------------------
// Detection scores + normalized features: one wave per pcd point (lane=channel)
// ---------------------------------------------------------------------------
__global__ __launch_bounds__(256) void detect_kernel(
    const float* fpf, const int* nbd, const float* gmaxp,
    float* out_scores, float* out_feat)
{
  int p    = (blockIdx.x * 256 + threadIdx.x) >> 5;
  int lane = threadIdx.x & 31;
  if (p >= N_PCDp) return;
  float inv  = 1.f / (gmaxp[0] + 1e-6f);
  float fraw = fpf[p * 32 + lane];
  float fn   = fraw * inv;
  float sum = 0.f;
  for (int j = 0; j < 64; ++j) {
    int nb = nbd[p * 64 + j];
    sum += fpf[nb * 32 + lane];
  }
  float mean = sum * inv * (1.f / 64.f);
  float x  = fn - mean;
  float lm = fmaxf(x, 0.f) + log1pf(expf(-fabsf(x)));   // softplus = logaddexp(x,0)
  float dmax = fn;
  for (int off = 16; off > 0; off >>= 1) dmax = fmaxf(dmax, __shfl_xor(dmax, off));
  float dw = fn / (1e-6f + dmax);
  float s  = lm * dw;
  float smax = s;
  for (int off = 16; off > 0; off >>= 1) smax = fmaxf(smax, __shfl_xor(smax, off));
  float n2 = fraw * fraw;
  for (int off = 16; off > 0; off >>= 1) n2 += __shfl_xor(n2, off);
  float nrm = fmaxf(sqrtf(n2), 1e-12f);
  out_feat[p * 32 + lane] = fraw / nrm;
  if (lane == 0) out_scores[p] = smax;
}

__global__ __launch_bounds__(256) void copy_kernel(const float* src, float* dst, int n)
{
  int i = blockIdx.x * 256 + threadIdx.x;
  if (i < n) dst[i] = src[i];
}

// ---------------------------------------------------------------------------
// Launcher
// ---------------------------------------------------------------------------
extern "C" void kernel_launch(void* const* d_in, const int* in_sizes, int n_in,
                              void* d_out, int out_size, void* d_ws, size_t ws_size,
                              hipStream_t stream)
{
  (void)in_sizes; (void)n_in; (void)out_size; (void)ws_size;
  const float* pcd_xyz  = (const float*)d_in[0];
  const float* pcd_feat = (const float*)d_in[1];
  const float* rgb_xyz  = (const float*)d_in[2];
  const float* rgb_feat = (const float*)d_in[3];
  const float* sa_W1 = (const float*)d_in[4];
  const float* sa_b1 = (const float*)d_in[5];
  const float* sa_W2 = (const float*)d_in[6];
  const float* sa_b2 = (const float*)d_in[7];
  const float* sa_W3 = (const float*)d_in[8];
  const float* sa_b3 = (const float*)d_in[9];
  const float* fp_W1 = (const float*)d_in[10];
  const float* fp_b1 = (const float*)d_in[11];
  const float* fp_W2 = (const float*)d_in[12];
  const float* fp_b2 = (const float*)d_in[13];
  float* out = (float*)d_out;

  char* ws = (char*)d_ws;
  size_t off = 0;
  auto carve = [&](size_t bytes) -> char* {
    char* p = ws + off;
    off += (bytes + 255) & ~(size_t)255;
    return p;
  };
  float*     XYZ    = (float*)    carve(N_ALLp * 3 * 4);
  float*     NEWXYZ = (float*)    carve(N_ANCH * 3 * 4);
  int*       NBSA   = (int*)      carve(N_ANCH * 32 * 4);
  float*     SAFEAT = (float*)    carve(N_ANCH * 128 * 4);
  _Float16*  W1H    = (_Float16*) carve(128 * 192 * 2);
  _Float16*  W2H    = (_Float16*) carve(128 * 128 * 2);
  _Float16*  W3H    = (_Float16*) carve(128 * 128 * 2);
  _Float16*  F1H    = (_Float16*) carve(128 * 160 * 2);
  _Float16*  F2H    = (_Float16*) carve(32 * 128 * 2);
  _Float16*  AFP    = (_Float16*) carve(5008 * 160 * 2);
  _Float16*  CFP    = (_Float16*) carve(5008 * 128 * 2);
  float*     FPFEAT = (float*)    carve(5008 * 32 * 4);
  int*       NBD    = (int*)      carve(N_PCDp * 64 * 4);
  float*     PART   = (float*)    carve(256 * 4);
  float*     GMAX   = (float*)    carve(4);

  // prep
  prep_weights_kernel<<<96, 256, 0, stream>>>(sa_W1, sa_W2, sa_W3, fp_W1, fp_W2,
                                              W1H, W2H, W3H, F1H, F2H);
  concat_xyz_kernel<<<176, 256, 0, stream>>>(pcd_xyz, rgb_xyz, XYZ);

  // FPS -> new_xyz
  fps_kernel<<<1, 1024, 0, stream>>>(XYZ, NEWXYZ);

  // SA ball query (4096 waves)
  ball_query_kernel<<<512, 256, 0, stream>>>(XYZ, NEWXYZ, N_ALLp, N_ANCH,
                                             0.1f * 0.1f, 32, NBSA);

  // SA MLP (WMMA) -> sa_feat
  sa_mlp_kernel<<<N_ANCH, 256, 0, stream>>>(XYZ, NEWXYZ, NBSA, pcd_feat, rgb_feat,
                                            W1H, W2H, W3H, sa_b1, sa_b2, sa_b3, SAFEAT);

  // FP 3-NN interpolation -> Afp (5008 waves)
  fp_interp_kernel<<<626, 256, 0, stream>>>(pcd_xyz, pcd_feat, NEWXYZ, SAFEAT, AFP);

  // FP layer1: (5008 x 160) @ (160 x 128)^T, 313*8 = 2504 waves
  fp_gemm_kernel<<<313, 256, 0, stream>>>(AFP, F1H, fp_b1, 160, 5, 8, 2504,
                                          CFP, nullptr, 128);
  // FP layer2: (5008 x 128) @ (128 x 32)^T, 313*2 = 626 waves
  fp_gemm_kernel<<<79, 256, 0, stream>>>(CFP, F2H, fp_b2, 128, 4, 2, 626,
                                         nullptr, FPFEAT, 32);

  // Detection ball query on pcd points (5000 waves)
  ball_query_kernel<<<625, 256, 0, stream>>>(pcd_xyz, pcd_xyz, N_PCDp, N_PCDp,
                                             0.2f * 0.2f, 64, NBD);

  // Global max of fp_feat[:5000]
  max_part_kernel<<<256, 256, 0, stream>>>(FPFEAT, N_PCDp * 32, PART);
  max_final_kernel<<<1, 256, 0, stream>>>(PART, 256, GMAX);

  // Scores + normalized features
  detect_kernel<<<625, 256, 0, stream>>>(FPFEAT, NBD, GMAX,
                                         out + N_PCDp * 3, out + N_PCDp * 3 + N_PCDp);

  // pcd_xyz passthrough
  copy_kernel<<<59, 256, 0, stream>>>(pcd_xyz, out, N_PCDp * 3);
}
